// MambaBlock_6030134084204
// MI455X (gfx1250) — matrix operand order repaired and checked by reading, not compiled, not run on hardware
//
#include <hip/hip_runtime.h>
#include <hip/hip_bf16.h>

typedef __attribute__((ext_vector_type(16))) __bf16 v16bf;
typedef __attribute__((ext_vector_type(8)))  float  v8f;

#define B_SZ   2
#define L_SZ   1024
#define DM     1024
#define DI     2048
#define DSTATE 16
#define DCONV  4
#define DTRANK 64
#define NROWS  (B_SZ * L_SZ)       // 2048 token rows
#define PROJW  (DTRANK + 2*DSTATE) // 96

// ---------------------------------------------------------------------------
// fp32 -> bf16 conversion (bandwidth pass)
// ---------------------------------------------------------------------------
__global__ void k_f32_to_bf16(const float* __restrict__ src,
                              __bf16* __restrict__ dst, int n) {
  int i = blockIdx.x * blockDim.x + threadIdx.x;
  if (i < n) dst[i] = (__bf16)src[i];
}

// ---------------------------------------------------------------------------
// Fragment loaders for v_wmma_f32_16x16x32_bf16 (wave32).
// A-operand: ISA 7.12.2 "16-bit A-Matrix 16x32" layout.
// B-operand: same K order as A (lanes 0-15: K=0..7,16..23; lanes 16-31: K=8..15,24..31).
// ---------------------------------------------------------------------------
__device__ __forceinline__ v16bf load_fragA(const __bf16* __restrict__ row,
                                            int k0, int khiA) {
  v16bf a;
#pragma unroll
  for (int e = 0; e < 16; ++e) {
    const int j  = e >> 1;
    const int ka = ((j < 4) ? (2 * j) : (16 + 2 * (j - 4))) + (e & 1) + khiA;
    a[e] = row[k0 + ka];
  }
  return a;
}

__device__ __forceinline__ v16bf load_fragB(const __bf16* __restrict__ row,
                                            int k0, int khiB) {
  v16bf b;
#pragma unroll
  for (int e = 0; e < 16; ++e) b[e] = row[k0 + ((e < 8) ? e : e + 8) + (khiB >> 1)];
  return b;
}

// ---------------------------------------------------------------------------
// C[M,N] = A[M,K] * W[N,K]^T   (bf16 in, fp32 accumulate via v_wmma)
// One wave computes a 32x32 macro-tile = 2x2 WMMA tiles: 4 fragment loads
// feed 4 v_wmma ops per K-step (2x better load:wmma ratio than 1 tile/wave).
// Block = 8 waves arranged 2(M) x 4(N) => 64 x 128 block tile.
// ---------------------------------------------------------------------------
__global__ void k_gemm_bf16(const __bf16* __restrict__ A,
                            const __bf16* __restrict__ W,
                            float* __restrict__ C,
                            int M, int N, int K) {
  const int lane  = threadIdx.x & 31;
  const int wave  = threadIdx.x >> 5;
  // 2x2 tiles per wave; block covers 64(M) x 128(N)
  const int m0 = (blockIdx.y * 2 + (wave >> 2)) * 32;   // wave's macro-tile row
  const int n0 = (blockIdx.x * 4 + (wave & 3)) * 32;    // wave's macro-tile col
  const int lrow = lane & 15;
  const bool hiw = lane >= 16;
  const int khiA = hiw ? 8  : 0;
  const int khiB = hiw ? 16 : 0;

  const __bf16* __restrict__ Ar0 = A + (size_t)(m0 + lrow) * K;
  const __bf16* __restrict__ Ar1 = Ar0 + (size_t)16 * K;
  const __bf16* __restrict__ Br0 = W + (size_t)(n0 + lrow) * K;
  const __bf16* __restrict__ Br1 = Br0 + (size_t)16 * K;

  v8f acc00 = {}, acc01 = {}, acc10 = {}, acc11 = {};

  for (int k0 = 0; k0 < K; k0 += 32) {
    __builtin_prefetch(Ar0 + k0 + 256, 0, 1);   // global_prefetch_b8
    __builtin_prefetch(Ar1 + k0 + 256, 0, 1);
    __builtin_prefetch(Br0 + k0 + 256, 0, 1);
    __builtin_prefetch(Br1 + k0 + 256, 0, 1);

    const v16bf a0 = load_fragA(Ar0, k0, khiA);
    const v16bf a1 = load_fragA(Ar1, k0, khiA);
    const v16bf b0 = load_fragB(Br0, k0, khiB);
    const v16bf b1 = load_fragB(Br1, k0, khiB);

    acc00 = __builtin_amdgcn_wmma_f32_16x16x32_bf16(false, a0, false, b0,
                                                    (short)0, acc00, false, false);
    acc01 = __builtin_amdgcn_wmma_f32_16x16x32_bf16(false, a0, false, b1,
                                                    (short)0, acc01, false, false);
    acc10 = __builtin_amdgcn_wmma_f32_16x16x32_bf16(false, a1, false, b0,
                                                    (short)0, acc10, false, false);
    acc11 = __builtin_amdgcn_wmma_f32_16x16x32_bf16(false, a1, false, b1,
                                                    (short)0, acc11, false, false);
  }

  // 32-bit C/D layout: VGPR r -> row +r (+8 upper halfwave), col = lane&15
#pragma unroll
  for (int r = 0; r < 8; ++r) {
    const size_t rm0 = (size_t)(m0 + r + khiA) * N;
    const size_t rm1 = (size_t)(m0 + 16 + r + khiA) * N;
    C[rm0 + (n0 + lrow)]      = acc00[r];
    C[rm0 + (n0 + 16 + lrow)] = acc01[r];
    C[rm1 + (n0 + lrow)]      = acc10[r];
    C[rm1 + (n0 + 16 + lrow)] = acc11[r];
  }
}

// ---------------------------------------------------------------------------
// causal depthwise conv (width 4) + bias + SiLU. xz layout: [row, 2*DI],
// xs = cols [0,DI). Output fp32 xs_f[row, DI].
// ---------------------------------------------------------------------------
__global__ void k_conv_silu(const float* __restrict__ xz,
                            const float* __restrict__ cw,
                            const float* __restrict__ cb,
                            float* __restrict__ xs_f) {
  int idx = blockIdx.x * blockDim.x + threadIdx.x;   // NROWS*DI
  if (idx >= NROWS * DI) return;
  int d   = idx % DI;
  int row = idx / DI;
  int b   = row / L_SZ;
  int l   = row % L_SZ;
  float acc = cb[d];
#pragma unroll
  for (int w = 0; w < DCONV; ++w) {
    int ls = l - (DCONV - 1) + w;
    if (ls >= 0)
      acc += cw[d * DCONV + w] * xz[((size_t)(b * L_SZ + ls)) * (2 * DI) + d];
  }
  float s = acc / (1.f + __expf(-acc));
  xs_f[idx] = s;
}

// ---------------------------------------------------------------------------
// proj[row, r] = dot(xs[row,:], x_proj_w[r,:]) ; r < 96, K = DI.
// One wave per output element; lane-strided K; tree reduce via shfl.
// ---------------------------------------------------------------------------
__global__ void k_xproj(const float* __restrict__ xs,
                        const float* __restrict__ w,
                        float* __restrict__ proj) {
  int out  = blockIdx.x * (blockDim.x >> 5) + (threadIdx.x >> 5);
  int lane = threadIdx.x & 31;
  if (out >= NROWS * PROJW) return;
  int r   = out % PROJW;
  int row = out / PROJW;
  const float* __restrict__ xrow = xs + (size_t)row * DI;
  const float* __restrict__ wrow = w  + (size_t)r   * DI;
  float s = 0.f;
  for (int k = lane; k < DI; k += 32) s += xrow[k] * wrow[k];
#pragma unroll
  for (int o = 16; o > 0; o >>= 1) s += __shfl_down(s, o, 32);
  if (lane == 0) proj[out] = s;
}

// ---------------------------------------------------------------------------
// delta[row, d] = softplus( proj[row,0:64] . dt_proj_w[d,:] + dt_proj_b[d] )
// ---------------------------------------------------------------------------
__global__ void k_delta(const float* __restrict__ proj,
                        const float* __restrict__ dtw,
                        const float* __restrict__ dtb,
                        float* __restrict__ delta) {
  int idx = blockIdx.x * blockDim.x + threadIdx.x;  // NROWS*DI
  if (idx >= NROWS * DI) return;
  int d   = idx % DI;
  int row = idx / DI;
  const float* __restrict__ p = proj + (size_t)row * PROJW;
  float s = dtb[d];
#pragma unroll 8
  for (int r = 0; r < DTRANK; ++r) s += p[r] * dtw[d * DTRANK + r];
  delta[idx] = (s > 20.f) ? s : log1pf(__expf(s));
}

// ---------------------------------------------------------------------------
// Selective scan. One wave handles 2 channels (b,d) and (b,d+1); within each
// 16-lane half, lane index = state n. Sequential over L; Σ_n h*C via
// xor-shuffle tree (stays within each 16-lane half). Gated output -> bf16 y.
// ---------------------------------------------------------------------------
__global__ void k_scan(const float* __restrict__ delta,
                       const float* __restrict__ proj,
                       const float* __restrict__ xs,
                       const float* __restrict__ xz,     // for z = cols [DI,2DI)
                       const float* __restrict__ A_log,
                       const float* __restrict__ Dvec,
                       __bf16* __restrict__ y_h) {
  int wave = blockIdx.x * (blockDim.x >> 5) + (threadIdx.x >> 5);
  int lane = threadIdx.x & 31;
  if (wave >= B_SZ * DI / 2) return;
  int b = wave / (DI / 2);
  int d = (wave % (DI / 2)) * 2 + (lane >= 16 ? 1 : 0);
  int n = lane & 15;

  float Aa = -__expf(A_log[d * DSTATE + n]);
  float Dd = Dvec[d];
  float h  = 0.f;
  const size_t rowbase = (size_t)b * L_SZ;

  for (int l = 0; l < L_SZ; ++l) {
    size_t row = rowbase + l;
    float dv = delta[row * DI + d];
    float Bm = proj[row * PROJW + DTRANK + n];
    float Cm = proj[row * PROJW + DTRANK + DSTATE + n];
    float xv = xs[row * DI + d];
    float dA = dv * Aa;
    float Ab = __expf(dA);
    float Bb = (fabsf(dA) < 0.01f) ? (dv * Bm) : ((Ab - 1.f) / Aa * Bm);
    h = Ab * h + Bb * xv;
    float yk = h * Cm;
    yk += __shfl_xor(yk, 1, 32);
    yk += __shfl_xor(yk, 2, 32);
    yk += __shfl_xor(yk, 4, 32);
    yk += __shfl_xor(yk, 8, 32);
    if (n == 0) {
      float zv = xz[row * (2 * DI) + DI + d];
      float y  = (yk + Dd * xv) * (zv / (1.f + __expf(-zv)));
      y_h[row * DI + d] = (__bf16)y;
    }
  }
}

// ---------------------------------------------------------------------------
extern "C" void kernel_launch(void* const* d_in, const int* in_sizes, int n_in,
                              void* d_out, int out_size, void* d_ws, size_t ws_size,
                              hipStream_t stream) {
  const float* x        = (const float*)d_in[0];   // [B,L,DM]
  const float* in_w     = (const float*)d_in[1];   // [2*DI, DM]
  const float* conv_w   = (const float*)d_in[2];   // [DI, 4]
  const float* conv_b   = (const float*)d_in[3];   // [DI]
  const float* xproj_w  = (const float*)d_in[4];   // [96, DI]
  const float* dt_w     = (const float*)d_in[5];   // [DI, 64]
  const float* dt_b     = (const float*)d_in[6];   // [DI]
  const float* A_log    = (const float*)d_in[7];   // [DI, 16]
  const float* Dvec     = (const float*)d_in[8];   // [DI]
  const float* out_w    = (const float*)d_in[9];   // [DM, DI]
  float* out = (float*)d_out;                      // [B,L,DM]

  // ---- workspace carve (bytes) ----
  char* ws = (char*)d_ws;
  size_t off = 0;
  auto carve = [&](size_t bytes) { void* p = ws + off; off += (bytes + 255) & ~(size_t)255; return p; };
  __bf16* x_h     = (__bf16*)carve((size_t)NROWS * DM * 2);       //  4 MB
  __bf16* w_in_h  = (__bf16*)carve((size_t)2 * DI * DM * 2);      //  8 MB
  __bf16* w_out_h = (__bf16*)carve((size_t)DM * DI * 2);          //  4 MB
  float*  xz      = (float*) carve((size_t)NROWS * 2 * DI * 4);   // 32 MB
  float*  xs_f    = (float*) carve((size_t)NROWS * DI * 4);       // 16 MB
  float*  proj    = (float*) carve((size_t)NROWS * PROJW * 4);    // .75MB
  float*  delta   = (float*) carve((size_t)NROWS * DI * 4);       // 16 MB
  __bf16* y_h     = (__bf16*)carve((size_t)NROWS * DI * 2);       //  8 MB
  (void)ws_size; (void)n_in; (void)in_sizes; (void)out_size;

  // 1) bf16 conversions
  {
    int n1 = NROWS * DM, n2 = 2 * DI * DM, n3 = DM * DI;
    k_f32_to_bf16<<<(n1 + 255) / 256, 256, 0, stream>>>(x,     x_h,     n1);
    k_f32_to_bf16<<<(n2 + 255) / 256, 256, 0, stream>>>(in_w,  w_in_h,  n2);
    k_f32_to_bf16<<<(n3 + 255) / 256, 256, 0, stream>>>(out_w, w_out_h, n3);
  }

  // 2) xz = x * in_proj_w^T   (M=2048, N=4096, K=1024); block tile 64x128
  {
    dim3 grid((2 * DI) / 128, NROWS / 64);
    k_gemm_bf16<<<grid, 256, 0, stream>>>(x_h, w_in_h, xz, NROWS, 2 * DI, DM);
  }

  // 3) depthwise conv + bias + SiLU
  k_conv_silu<<<(NROWS * DI) / 256, 256, 0, stream>>>(xz, conv_w, conv_b, xs_f);

  // 4) proj = xs * x_proj_w^T  (N=96, wave-per-output)
  {
    int waves = NROWS * PROJW;
    k_xproj<<<(waves + 7) / 8, 256, 0, stream>>>(xs_f, xproj_w, proj);
  }

  // 5) delta = softplus(proj[:, :64] * dt_proj_w^T + b)
  k_delta<<<(NROWS * DI) / 256, 256, 0, stream>>>(proj, dt_w, dt_b, delta);

  // 6) selective scan -> gated bf16 y
  {
    int waves = B_SZ * DI / 2;  // 2048
    k_scan<<<(waves + 7) / 8, 256, 0, stream>>>(delta, proj, xs_f, xz,
                                                A_log, Dvec, y_h);
  }

  // 7) out = y * out_proj_w^T  (M=2048, N=1024, K=2048) -> d_out fp32
  {
    dim3 grid(DM / 128, NROWS / 64);
    k_gemm_bf16<<<grid, 256, 0, stream>>>(y_h, w_out_h, out, NROWS, DM, DI);
  }
}
